// CKAN_21096879358344
// MI455X (gfx1250) — matrix operand order repaired and checked by reading, not compiled, not run on hardware
//
#include <hip/hip_runtime.h>
#include <math.h>

typedef __attribute__((ext_vector_type(16))) _Float16 v16h;
typedef __attribute__((ext_vector_type(8)))  _Float16 v8h;
typedef __attribute__((ext_vector_type(4)))  _Float16 v4h;
typedef __attribute__((ext_vector_type(8)))  float    v8f;

#define DIM_   64
#define B_SZ   2048
#define M_SZ   64
#define BM_    (B_SZ * M_SZ)

__device__ __forceinline__ float sigmoidf_(float x) { return 1.f / (1.f + __expf(-x)); }

// ---------------------------------------------------------------------------
// Convert weight matrices to f16 row-major [out][in]; this is exactly the
// contiguous per-lane k layout this code uses for WMMA's B operand (16 K-contig halves/lane).
// ---------------------------------------------------------------------------
__global__ void k_cvt_weights(const float* __restrict__ w1, const float* __restrict__ w2,
                              const float* __restrict__ wagg,
                              _Float16* __restrict__ w1h, _Float16* __restrict__ w2h,
                              _Float16* __restrict__ waggh)
{
    int i = blockIdx.x * 256 + threadIdx.x;
    if (i < 64 * 128) w1h[i]   = (_Float16)w1[i];
    if (i < 64 * 64)  w2h[i]   = (_Float16)w2[i];
    if (i < 64 * 192) waggh[i] = (_Float16)wagg[i];
}

// ---------------------------------------------------------------------------
// Hop-0: e0[b,:] = mean_m table[ent[b,m],:]   (writes column 0..63 of emb[b,192])
// ---------------------------------------------------------------------------
__global__ void k_hop0_mean(const float* __restrict__ table, const int* __restrict__ ent,
                            float* __restrict__ embcol)
{
    __shared__ float part[256];
    const int b  = blockIdx.x;
    const int d  = threadIdx.x & 63;
    const int mq = threadIdx.x >> 6;
    float acc = 0.f;
    for (int m = mq * 16; m < mq * 16 + 16; ++m) {
        int idx = ent[b * M_SZ + m];
        acc += table[(size_t)idx * DIM_ + d];
    }
    part[threadIdx.x] = acc;
    __syncthreads();
    if (mq == 0)
        embcol[(size_t)b * 192 + d] =
            (part[d] + part[64 + d] + part[128 + d] + part[192 + d]) * (1.f / 64.f);
}

// ---------------------------------------------------------------------------
// DNN scorer: per wave, 16 rows of X=[head_emb|rel_emb] (16x128 f16 in LDS),
// Y1 = relu(X@w1^T+b1) via 4x4 WMMA, Y2 = relu(Y1@w2^T+b2) via 2x4 WMMA,
// logit = Y2.w3 + b3 (lane-shuffle reduce), store sigmoid(logit).
// ---------------------------------------------------------------------------
__global__ __launch_bounds__(256) void k_dnn_logits(
    const float* __restrict__ ent_emb, const float* __restrict__ rel_emb,
    const int* __restrict__ heads, const int* __restrict__ rels,
    const _Float16* __restrict__ w1h, const _Float16* __restrict__ w2h,
    const float* __restrict__ w3, const float* __restrict__ b1,
    const float* __restrict__ b2, const float* __restrict__ b3,
    float* __restrict__ logits)
{
    __shared__ _Float16 ldsX[8][16][128];
    __shared__ _Float16 ldsY[8][16][64];
    const int tid  = threadIdx.x;
    const int wave = tid >> 5, lane = tid & 31;
    const int row  = lane & 15, half = lane >> 4;
    const int r0   = blockIdx.x * 128 + wave * 16;

    // ---- stage X rows into LDS as f16 (half 0: head emb, half 1: rel emb) ----
    {
        const int gr = r0 + row;
        const float* src = half ? (rel_emb + (size_t)rels[gr] * DIM_)
                                : (ent_emb + (size_t)heads[gr] * DIM_);
        _Float16* dst = &ldsX[wave][row][half * 64];
        const float4* s4 = (const float4*)src;
#pragma unroll
        for (int j = 0; j < 16; ++j) {
            float4 v = s4[j];
            v4h h = { (_Float16)v.x, (_Float16)v.y, (_Float16)v.z, (_Float16)v.w };
            *(v4h*)(dst + 4 * j) = h;
        }
    }
    __syncthreads();

    const int col = row;                  // lane&15 = output column within tile
    const int kb8 = half * 8, kb16 = half * 16;

    // ---- GEMM1: acc = X[16x128] @ w1^T[128x64], K-tiles of 32 ----
    v8f acc[4] = {};
#pragma unroll
    for (int kt = 0; kt < 4; ++kt) {
        const _Float16* xr = &ldsX[wave][row][kt * 32];
        v8h a0 = *(const v8h*)(xr + kb16);
        v8h a1 = *(const v8h*)(xr + kb16 + 8);
        union { v16h v; v8h h[2]; } au; au.h[0] = a0; au.h[1] = a1;
#pragma unroll
        for (int nt = 0; nt < 4; ++nt) {
            v16h bfrag = *(const v16h*)(w1h + (nt * 16 + col) * 128 + kt * 32 + kb16);
            acc[nt] = __builtin_amdgcn_wmma_f32_16x16x32_f16(
                false, au.v, false, bfrag, (short)0, acc[nt], false, false);
        }
    }
    // bias + relu, back to LDS as f16 (D layout: row=(lane>>4)*8+i, col=lane&15)
#pragma unroll
    for (int nt = 0; nt < 4; ++nt) {
        const int cg = nt * 16 + col;
        const float bb = b1[cg];
#pragma unroll
        for (int i = 0; i < 8; ++i) {
            float y = fmaxf(acc[nt][i] + bb, 0.f);
            ldsY[wave][half * 8 + i][cg] = (_Float16)y;
        }
    }
    __syncthreads();

    // ---- GEMM2 (K=64) fused with logit dot against w3 ----
    v8f acc2[4] = {};
#pragma unroll
    for (int kt = 0; kt < 2; ++kt) {
        const _Float16* yr = &ldsY[wave][row][kt * 32];
        v8h a0 = *(const v8h*)(yr + kb16);
        v8h a1 = *(const v8h*)(yr + kb16 + 8);
        union { v16h v; v8h h[2]; } au; au.h[0] = a0; au.h[1] = a1;
#pragma unroll
        for (int nt = 0; nt < 4; ++nt) {
            v16h bfrag = *(const v16h*)(w2h + (nt * 16 + col) * 64 + kt * 32 + kb16);
            acc2[nt] = __builtin_amdgcn_wmma_f32_16x16x32_f16(
                false, au.v, false, bfrag, (short)0, acc2[nt], false, false);
        }
    }
    float p[8] = {0.f, 0.f, 0.f, 0.f, 0.f, 0.f, 0.f, 0.f};
#pragma unroll
    for (int nt = 0; nt < 4; ++nt) {
        const int cg = nt * 16 + col;
        const float bb = b2[cg];
        const float wc = w3[cg];
#pragma unroll
        for (int i = 0; i < 8; ++i) {
            float y = fmaxf(acc2[nt][i] + bb, 0.f);
            p[i] += y * wc;
        }
    }
    // reduce over 16 columns (xor shuffles stay within each 16-lane half)
#pragma unroll
    for (int i = 0; i < 8; ++i) {
#pragma unroll
        for (int off = 1; off < 16; off <<= 1)
            p[i] += __shfl_xor(p[i], off, 32);
    }
    if (col == 0) {
        const float b3v = b3[0];
#pragma unroll
        for (int i = 0; i < 8; ++i)
            logits[r0 + half * 8 + i] = sigmoidf_(p[i] + b3v);
    }
}

// ---------------------------------------------------------------------------
// softmax over M neighbors (of sigmoid scores) + weighted tail-embedding sum.
// One block per b, 64 threads; thread m also owns output dim d = m.
// ---------------------------------------------------------------------------
__global__ void k_soft_agg(const float* __restrict__ s, const int* __restrict__ tails,
                           const float* __restrict__ ent_emb, float* __restrict__ embcol)
{
    __shared__ float pi[64];
    __shared__ float red[64];
    __shared__ int   ti[64];
    const int b = blockIdx.x, m = threadIdx.x;
    float v = s[b * M_SZ + m];
    ti[m]  = tails[b * M_SZ + m];
    pi[m]  = v;
    red[m] = v;
    __syncthreads();
    for (int off = 32; off > 0; off >>= 1) {
        if (m < off) red[m] = fmaxf(red[m], red[m + off]);
        __syncthreads();
    }
    const float mx = red[0];
    __syncthreads();
    float e = __expf(v - mx);
    pi[m] = e; red[m] = e;
    __syncthreads();
    for (int off = 32; off > 0; off >>= 1) {
        if (m < off) red[m] += red[m + off];
        __syncthreads();
    }
    const float inv = 1.f / red[0];
    const int d = m;
    float acc = 0.f;
    for (int mm = 0; mm < M_SZ; ++mm)
        acc += pi[mm] * ent_emb[(size_t)ti[mm] * DIM_ + d];   // coalesced row reads
    embcol[(size_t)b * 192 + d] = acc * inv;
}

// ---------------------------------------------------------------------------
// Aggregation: out[b,:] = sigmoid(emb[b,192] @ wagg^T + bagg) via 6x4 WMMA.
// ---------------------------------------------------------------------------
__global__ __launch_bounds__(256) void k_agg(
    const float* __restrict__ emb, const _Float16* __restrict__ waggh,
    const float* __restrict__ bagg, float* __restrict__ outv)
{
    __shared__ _Float16 ldsE[8][16][192];
    const int tid  = threadIdx.x;
    const int wave = tid >> 5, lane = tid & 31;
    const int row  = lane & 15, half = lane >> 4;
    const int r0   = blockIdx.x * 128 + wave * 16;
    {
        const float* src = emb + (size_t)(r0 + row) * 192 + half * 96;
        _Float16* dst = &ldsE[wave][row][half * 96];
        const float4* s4 = (const float4*)src;
#pragma unroll
        for (int j = 0; j < 24; ++j) {
            float4 v = s4[j];
            v4h h = { (_Float16)v.x, (_Float16)v.y, (_Float16)v.z, (_Float16)v.w };
            *(v4h*)(dst + 4 * j) = h;
        }
    }
    __syncthreads();
    const int col = row;
    const int kb8 = half * 8, kb16 = half * 16;
    v8f acc[4] = {};
#pragma unroll
    for (int kt = 0; kt < 6; ++kt) {
        const _Float16* xr = &ldsE[wave][row][kt * 32];
        v8h a0 = *(const v8h*)(xr + kb16);
        v8h a1 = *(const v8h*)(xr + kb16 + 8);
        union { v16h v; v8h h[2]; } au; au.h[0] = a0; au.h[1] = a1;
#pragma unroll
        for (int nt = 0; nt < 4; ++nt) {
            v16h bfrag = *(const v16h*)(waggh + (nt * 16 + col) * 192 + kt * 32 + kb16);
            acc[nt] = __builtin_amdgcn_wmma_f32_16x16x32_f16(
                false, au.v, false, bfrag, (short)0, acc[nt], false, false);
        }
    }
#pragma unroll
    for (int nt = 0; nt < 4; ++nt) {
        const int cg = nt * 16 + col;
        const float bb = bagg[cg];
#pragma unroll
        for (int i = 0; i < 8; ++i)
            outv[(size_t)(r0 + half * 8 + i) * DIM_ + cg] = sigmoidf_(acc[nt][i] + bb);
    }
}

// ---------------------------------------------------------------------------
// Final: out[b] = sigmoid(users[b,:] . items[b,:])
// ---------------------------------------------------------------------------
__global__ void k_final(const float* __restrict__ users, const float* __restrict__ items,
                        float* __restrict__ out)
{
    int b = blockIdx.x * blockDim.x + threadIdx.x;
    if (b >= B_SZ) return;
    float s = 0.f;
    for (int d = 0; d < DIM_; ++d)
        s += users[b * DIM_ + d] * items[b * DIM_ + d];
    out[b] = sigmoidf_(s);
}

extern "C" void kernel_launch(void* const* d_in, const int* in_sizes, int n_in,
                              void* d_out, int out_size, void* d_ws, size_t ws_size,
                              hipStream_t stream)
{
    (void)in_sizes; (void)n_in; (void)out_size; (void)ws_size;
    const float* ent_emb = (const float*)d_in[0];
    const float* rec_emb = (const float*)d_in[1];
    const float* rel_emb = (const float*)d_in[2];
    const float* w1   = (const float*)d_in[3];
    const float* b1   = (const float*)d_in[4];
    const float* w2   = (const float*)d_in[5];
    const float* b2   = (const float*)d_in[6];
    const float* w3   = (const float*)d_in[7];
    const float* b3   = (const float*)d_in[8];
    const float* wagg = (const float*)d_in[9];
    const float* bagg = (const float*)d_in[10];
    const int* u_ent   = (const int*)d_in[11];
    const int* u_heads = (const int*)d_in[12];
    const int* u_rels  = (const int*)d_in[13];
    const int* u_tails = (const int*)d_in[14];
    const int* v_ent   = (const int*)d_in[15];
    const int* v_heads = (const int*)d_in[16];
    const int* v_rels  = (const int*)d_in[17];
    const int* v_tails = (const int*)d_in[18];

    char* ws = (char*)d_ws;
    _Float16* w1h   = (_Float16*)(ws + 0);        //  16 KB
    _Float16* w2h   = (_Float16*)(ws + 16384);    //   8 KB
    _Float16* waggh = (_Float16*)(ws + 24576);    //  24 KB
    float* logits   = (float*)(ws + 49152);       // 512 KB (reused per hop/side)
    float* emb_u    = (float*)(ws + 573440);      // 1.5 MB [B][192]
    float* emb_v    = (float*)(ws + 2146304);     // 1.5 MB
    float* users    = (float*)(ws + 3719168);     // 512 KB [B][64]
    float* items    = (float*)(ws + 4243456);     // 512 KB

    k_cvt_weights<<<48, 256, 0, stream>>>(w1, w2, wagg, w1h, w2h, waggh);

    k_hop0_mean<<<B_SZ, 256, 0, stream>>>(rec_emb, u_ent, emb_u);
    k_hop0_mean<<<B_SZ, 256, 0, stream>>>(ent_emb, v_ent, emb_v);

    for (int l = 0; l < 2; ++l) {
        k_dnn_logits<<<BM_ / 128, 256, 0, stream>>>(ent_emb, rel_emb,
            u_heads + l * BM_, u_rels + l * BM_, w1h, w2h, w3, b1, b2, b3, logits);
        k_soft_agg<<<B_SZ, 64, 0, stream>>>(logits, u_tails + l * BM_, ent_emb,
                                            emb_u + (l + 1) * 64);
    }
    for (int l = 0; l < 2; ++l) {
        k_dnn_logits<<<BM_ / 128, 256, 0, stream>>>(ent_emb, rel_emb,
            v_heads + l * BM_, v_rels + l * BM_, w1h, w2h, w3, b1, b2, b3, logits);
        k_soft_agg<<<B_SZ, 64, 0, stream>>>(logits, v_tails + l * BM_, ent_emb,
                                            emb_v + (l + 1) * 64);
    }

    k_agg<<<16, 256, 0, stream>>>(emb_u, waggh, bagg, users);
    k_agg<<<16, 256, 0, stream>>>(emb_v, waggh, bagg, items);
    k_final<<<8, 256, 0, stream>>>(users, items, (float*)d_out);
}
